// RNN_47072841564417
// MI455X (gfx1250) — matrix operand order repaired and checked by reading, not compiled, not run on hardware
//
#include <hip/hip_runtime.h>
#include <hip/hip_bf16.h>

typedef __attribute__((ext_vector_type(16))) _Float16 v16h;
typedef __attribute__((ext_vector_type(8)))  _Float16 v8h;
typedef __attribute__((ext_vector_type(8)))  float    v8f;
typedef __attribute__((ext_vector_type(4)))  float    v4f;

#define NWG     32
#define T_STEPS 1024
#define BATCH   64
#define IN_DIM  256
#define HID     512
#define K_DIM   768           // IN_DIM + HID
#define N_DIM   2048          // 4 * HID
#define KCHUNKS 24            // K_DIM / 32
#define NTILES  128           // N_DIM / 16
#define A_STRIDE 776          // K_DIM + 8 halves pad: 388 dwords == 4 (mod 64) bank skew

// workspace layout (bytes)
#define BAR_OFF    0                  // u32 barrier counter
#define H16_OFF    4096               // f16 [64][512] hidden state    (64 KB)
#define WPACK_OFF  262144             // f16 packed W fragments        (3 MB)
#define X16_OFF    4194304            // f16 [64][1024][256] x in f16  (32 MB, optional)
#define X16_BYTES  ((size_t)BATCH * T_STEPS * IN_DIM * 2)
#define WS_NEED_X16 (X16_OFF + X16_BYTES)

__device__ __forceinline__ float sigm(float v) {
    return 1.0f / (1.0f + __expf(-v));
}
__device__ __forceinline__ float tanh_f(float v) {
    return 1.0f - 2.0f / (__expf(2.0f * v) + 1.0f);
}

// Async global->LDS 16B copy (ASYNCcnt path, CDNA5 ISA).
__device__ __forceinline__ void async_b128(unsigned lds_byte_off, const void* gptr) {
    unsigned long long ga = (unsigned long long)(uintptr_t)gptr;
    asm volatile("global_load_async_to_lds_b128 %0, %1, off"
                 :: "v"(lds_byte_off), "v"(ga) : "memory");
}
__device__ __forceinline__ void wait_async0() {
    asm volatile("s_wait_asynccnt 0x0" ::: "memory");
}

// Monotonic-counter grid barrier; init kernel zeroes the counter each call.
__device__ __forceinline__ void grid_sync(unsigned* bar) {
    __syncthreads();
    if (threadIdx.x == 0) {
        __threadfence();
        unsigned arrive =
            __hip_atomic_fetch_add(bar, 1u, __ATOMIC_ACQ_REL, __HIP_MEMORY_SCOPE_AGENT) + 1u;
        unsigned goal = ((arrive + NWG - 1u) / NWG) * NWG;
        while (__hip_atomic_load(bar, __ATOMIC_ACQUIRE, __HIP_MEMORY_SCOPE_AGENT) < goal) {
            __builtin_amdgcn_s_sleep(1);
        }
        __threadfence();
    }
    __syncthreads();
}

// Init: reset barrier, zero h16, pack W into per-lane B-fragment layout
// (lane = N col, half h -> K = kc*32 + 16*(h>=8) + 8*(lane>=16) + h%8), optionally convert
// x to f16 so the hot loop's staging is pure async B128 copies.
__global__ __launch_bounds__(256) void lstm_init(
    const float* __restrict__ W, const float* __restrict__ x,
    _Float16* __restrict__ wpack, _Float16* __restrict__ x16,
    _Float16* __restrict__ h16, unsigned* __restrict__ bar, int do_x16)
{
    size_t gid    = (size_t)blockIdx.x * blockDim.x + threadIdx.x;
    size_t stride = (size_t)gridDim.x * blockDim.x;
    if (gid == 0) *bar = 0u;
    for (size_t e = gid; e < (size_t)BATCH * HID; e += stride)
        h16[e] = (_Float16)0.0f;
    for (size_t e = gid; e < (size_t)NTILES * KCHUNKS * 32 * 16; e += stride) {
        int h       = (int)(e & 15);
        int lane    = (int)((e >> 4) & 31);
        size_t rest = e >> 9;                 // ntile*KCHUNKS + kc
        int kc      = (int)(rest % KCHUNKS);
        int ntile   = (int)(rest / KCHUNKS);
        int n = ntile * 16 + (lane & 15);
        int k = kc * 32 + ((h & 8) << 1) + ((lane >> 4) << 3) + (h & 7);
        wpack[e] = (_Float16)W[(size_t)k * N_DIM + n];
    }
    if (do_x16) {
        for (size_t e = gid; e < (size_t)BATCH * T_STEPS * IN_DIM; e += stride)
            x16[e] = (_Float16)x[e];
    }
}

// Persistent LSTM: 32 WGs x 256 threads; WG owns 16 hidden columns.
// Per step: async-stage h into private LDS (x_{t+1} was prefetched behind the
// previous barrier), 64x64 WMMA GEMM slice (4 gates x 16 cols, 8 waves x 2
// tiles), register-resident cell update, ONE grid sync.
__global__ __launch_bounds__(256) void lstm_persistent(
    const float* __restrict__ x, const float* __restrict__ bias,
    const _Float16* __restrict__ wpack, const _Float16* __restrict__ x16,
    _Float16* __restrict__ h16, unsigned* __restrict__ bar,
    float* __restrict__ out, int use_x16)
{
    const int wg    = blockIdx.x;            // hidden cols [wg*16, wg*16+16)
    const int tid   = threadIdx.x;
    const int wave  = tid >> 5;
    const int lane  = tid & 31;
    const int g     = wave & 3;              // gate (i,j,f,o)
    const int mt2   = (wave >> 2) << 1;      // m-tile base: 0 or 2
    const int ntile = g * 32 + wg;
    const int hcol0 = wg * 16;
    const int lhi   = lane >> 4;
    const int llo   = lane & 15;

    __shared__ _Float16 a_tile[BATCH][A_STRIDE];   // [x_t ++ h] staging, ~97 KB
    __shared__ float    gbuf[BATCH][65];           // gate staging, ~16.6 KB

    const unsigned abase = (unsigned)(uintptr_t)(&a_tile[0][0]);

    // A-fragment pointers (16-bit A 16x32 layout): lane llo = row,
    // halves 0..7 at K 8*lhi, halves 8..15 at K 16+8*lhi.
    const _Float16* arow0 = &a_tile[mt2 * 16 + llo][lhi << 3];
    const _Float16* arow1 = arow0 + 16 * A_STRIDE;
    const _Float16* wbase = wpack + ((size_t)ntile * KCHUNKS * 32 + lane) * 16;

    // Phase-C ownership: thread tid owns (m0 + 16i, hcol0 + hc), i = 0..3.
    const int hc    = tid & 15;
    const int m0    = tid >> 4;
    const int hcolC = hcol0 + hc;
    const float bi = bias[hcolC];
    const float bj = bias[HID + hcolC];
    const float bf2 = bias[2 * HID + hcolC];
    const float bo = bias[3 * HID + hcolC];
    float creg[4] = {0.0f, 0.0f, 0.0f, 0.0f};      // cell state lives in VGPRs

    // Stage x_t into the LDS A tile (async if f16 copy exists).
    auto stage_x = [&](int t) {
        if (use_x16) {
            for (int e = tid; e < BATCH * (IN_DIM / 8); e += 256) {
                int m = e >> 5, ch = e & 31;
                async_b128(abase + (unsigned)(m * A_STRIDE + ch * 8) * 2,
                           x16 + ((size_t)m * T_STEPS + t) * IN_DIM + ch * 8);
            }
        } else {
            for (int e = tid; e < BATCH * (IN_DIM / 4); e += 256) {
                int m = e >> 6, ch = e & 63;
                v4f v = *(const v4f*)(x + ((size_t)m * T_STEPS + t) * IN_DIM + ch * 4);
                _Float16* d = &a_tile[m][ch * 4];
                d[0] = (_Float16)v.x; d[1] = (_Float16)v.y;
                d[2] = (_Float16)v.z; d[3] = (_Float16)v.w;
            }
        }
    };

    stage_x(0);                                    // prologue: x_0

    for (int t = 0; t < T_STEPS; ++t) {
        // ---- Phase A: broadcast h into private LDS (x already in flight) ----
        for (int e = tid; e < BATCH * (HID / 8); e += 256) {
            int m = e >> 6, ch = e & 63;
            async_b128(abase + (unsigned)(m * A_STRIDE + IN_DIM + ch * 8) * 2,
                       h16 + m * HID + ch * 8);
        }
        wait_async0();
        __syncthreads();

        // ---- Phase B: GEMM slice, K = 768 in 24 chunks of 32 ----
        v8f acc0 = {}; v8f acc1 = {};
        const _Float16* wp = wbase;
        #pragma unroll 4
        for (int kc = 0; kc < KCHUNKS; ++kc) {
            v16h bf   = *(const v16h*)wp;                     // packed B frag, 32B
            v8h  a0lo = *(const v8h*)(arow0 + kc * 32);
            v8h  a0hi = *(const v8h*)(arow0 + kc * 32 + 16);
            v8h  a1lo = *(const v8h*)(arow1 + kc * 32);
            v8h  a1hi = *(const v8h*)(arow1 + kc * 32 + 16);
            v16h af0 = __builtin_shufflevector(a0lo, a0hi,
                        0,1,2,3,4,5,6,7,8,9,10,11,12,13,14,15);
            v16h af1 = __builtin_shufflevector(a1lo, a1hi,
                        0,1,2,3,4,5,6,7,8,9,10,11,12,13,14,15);
            acc0 = __builtin_amdgcn_wmma_f32_16x16x32_f16(
                       false, af0, false, bf, (short)0, acc0, false, false);
            acc1 = __builtin_amdgcn_wmma_f32_16x16x32_f16(
                       false, af1, false, bf, (short)0, acc1, false, false);
            wp += 32 * 16;
        }

        // Stage gate tiles: C/D layout lane->N=llo, VGPR r -> M=r+8*lhi.
        const int col = g * 16 + llo;
        const int rb  = lhi * 8;
        #pragma unroll
        for (int r = 0; r < 8; ++r) {
            gbuf[mt2 * 16 + rb + r][col]      = acc0[r];
            gbuf[mt2 * 16 + 16 + rb + r][col] = acc1[r];
        }
        __syncthreads();                 // a_tile fully consumed from here on

        // Prefetch x_{t+1}: overlaps Phase C and the grid barrier wait.
        if (t + 1 < T_STEPS) stage_x(t + 1);

        // ---- Phase C: register-resident cell update ----
        #pragma unroll
        for (int i = 0; i < 4; ++i) {
            int m = m0 + (i << 4);
            float iv = gbuf[m][hc]      + bi;
            float jv = gbuf[m][16 + hc] + bj;
            float fv = gbuf[m][32 + hc] + bf2;
            float ov = gbuf[m][48 + hc] + bo;
            float cn = sigm(fv + 1.0f) * creg[i] + sigm(iv) * tanh_f(jv);
            float hn = sigm(ov) * tanh_f(cn);
            creg[i] = cn;
            out[((size_t)m * T_STEPS + t) * HID + hcolC] = hn;
            h16[m * HID + hcolC] = (_Float16)hn;
        }
        grid_sync(bar);   // publish h16 before anyone stages step t+1's h
    }
}

extern "C" void kernel_launch(void* const* d_in, const int* in_sizes, int n_in,
                              void* d_out, int out_size, void* d_ws, size_t ws_size,
                              hipStream_t stream) {
    const float* x    = (const float*)d_in[0];   // [64,1024,256]
    const float* W    = (const float*)d_in[1];   // [768,2048]
    const float* bias = (const float*)d_in[2];   // [2048]
    float* out        = (float*)d_out;           // [64,1024,512]

    char* ws = (char*)d_ws;
    unsigned* bar     = (unsigned*)(ws + BAR_OFF);
    _Float16* h16     = (_Float16*)(ws + H16_OFF);
    _Float16* wpack   = (_Float16*)(ws + WPACK_OFF);
    _Float16* x16     = (_Float16*)(ws + X16_OFF);

    const int use_x16 = (ws_size >= WS_NEED_X16) ? 1 : 0;

    lstm_init<<<4096, 256, 0, stream>>>(W, x, wpack, x16, h16, bar, use_x16);
    lstm_persistent<<<NWG, 256, 0, stream>>>(x, bias, wpack, x16, h16,
                                             bar, out, use_x16);
}
